// Rotator_10746008175492
// MI455X (gfx1250) — matrix operand order repaired and checked by reading, not compiled, not run on hardware
//
#include <hip/hip_runtime.h>
#include <hip/hip_bf16.h>

typedef __bf16 bf16_t;
typedef bf16_t v16bf __attribute__((ext_vector_type(16)));
typedef bf16_t v8bf  __attribute__((ext_vector_type(8)));
typedef float  v8f   __attribute__((ext_vector_type(8)));

#define DIM    128
#define HEADS  8
#define DEPTH  3
#define TT     256      // len(batch_sizes)
#define TM     255      // scan steps
#define BTOT   512      // batch_sizes[0]
#define BTILE  16       // batch rows per workgroup (WMMA M)
#define NWG    (BTOT / BTILE)
#define LN_EPS 1e-5f

// ---------------------------------------------------------------------------
// Prep: prefix-sum of batch_sizes (grid gather offsets) + Theta fp32 -> bf16,
// transposed to [l][h][n][m] so WMMA B-fragments are contiguous along K=m.
// ---------------------------------------------------------------------------
__global__ void prep_kernel(const int* __restrict__ bs,
                            const float* __restrict__ Theta,
                            int* __restrict__ off,
                            bf16_t* __restrict__ thT) {
  if (blockIdx.x == 0 && threadIdx.x == 0) {
    int acc = 0;
    for (int i = 0; i < TT; ++i) { off[i] = acc; acc += bs[i]; }
    off[TT] = acc;
  }
  const int total = DEPTH * HEADS * DIM * DIM;
  for (int idx = blockIdx.x * blockDim.x + threadIdx.x; idx < total;
       idx += gridDim.x * blockDim.x) {
    int m  = idx & (DIM - 1);
    int n  = (idx >> 7) & (DIM - 1);
    int lh = idx >> 14;
    thT[idx] = (bf16_t)Theta[(lh * DIM + m) * DIM + n];
  }
}

// ---------------------------------------------------------------------------
// Main recurrent kernel: 1 block per 16 batch rows, wave h <-> head h.
// ---------------------------------------------------------------------------
__global__ __launch_bounds__(256, 1)
void rotator_kernel(const int* __restrict__ data, const int* __restrict__ bs,
                    const float* __restrict__ angle_emb,
                    const float* __restrict__ emb_real,
                    const float* __restrict__ emb_imag,
                    const float* __restrict__ proportion,
                    const float* __restrict__ lima_shape,
                    const float* __restrict__ ln_w, const float* __restrict__ ln_b,
                    const float* __restrict__ init_history,
                    const float* __restrict__ head_clock_denoms,
                    const float* __restrict__ rotary_denom_p,
                    const int* __restrict__ off, const bf16_t* __restrict__ thT,
                    float* __restrict__ out) {
  __shared__ float  s_theta[BTILE][DIM];
  __shared__ float  s_accum[BTILE][DIM];
  __shared__ bf16_t s_o[HEADS][BTILE][DIM];   // per-wave A-operand staging (bf16)
  __shared__ int    s_tok[BTILE];
  __shared__ int    s_lab[BTILE];
  __shared__ float  s_mask[BTILE];
  __shared__ int    s_zero;                   // anti-hoist token (always 0)

  const int tid     = threadIdx.x;
  const int h       = tid >> 5;     // wave == head
  const int lane    = tid & 31;
  const int lane_lo = lane & 15;
  const int half    = lane >> 4;
  const int bbase   = blockIdx.x * BTILE;

  if (tid == 0) s_zero = 0;

  // Per-(layer, n-tile) parameters at feature f = 16*t8 + lane_lo.
  float p_ba[DEPTH][8], p_scc[DEPTH][8], p_lnw[DEPTH][8], p_lnb[DEPTH][8];
  float hclk[8];
  v8f   hist[DEPTH][8];             // recurrent state in WMMA C layout

  {
    const float lden = __logf(head_clock_denoms[h]);
#pragma unroll
    for (int t8 = 0; t8 < 8; ++t8) {
      const int f = 16 * t8 + lane_lo;
      hclk[t8] = __expf(-(float)f * (1.0f / DIM) * lden);  // denom^(-f/dim)
#pragma unroll
      for (int l = 0; l < DEPTH; ++l) {
        const float ls = lima_shape[(l * HEADS + h) * DIM + f];
        const float pr = proportion[(l * HEADS + h) * DIM + f];
        p_ba[l][t8]  = (1.0f / (1.0f + __expf(-ls)) - 0.5f) * 0.5f;
        p_scc[l][t8] = 0.5f / (__expf(pr) + 1.0f);
        p_lnw[l][t8] = ln_w[l * DIM + f];
        p_lnb[l][t8] = ln_b[l * DIM + f];
        const float iv = init_history[(l * HEADS + h) * DIM + f];
#pragma unroll
        for (int r = 0; r < 8; ++r) hist[l][t8][r] = iv;
      }
    }
  }

  // Output-phase constants: thread handles row ob, 8 features starting at onb.
  const int ob  = tid >> 4;
  const int onb = (tid & 15) * 8;
  float tang[8];
  {
    const float lr = __logf(rotary_denom_p[0]);
#pragma unroll
    for (int i = 0; i < 8; ++i)
      tang[i] = __expf(-(float)(onb + i) * (1.0f / DIM) * lr);
  }

  for (int t = 0; t < TM; ++t) {
    // ---- gather tokens / labels / mask ----
    if (tid < BTILE) {
      const int bcol = bbase + tid;
      const int bst  = bs[t];
      const int bst1 = bs[t + 1];
      s_tok[tid]  = data[off[t]     + (bcol < bst  ? bcol : bst  - 1)];
      s_lab[tid]  = data[off[t + 1] + (bcol < bst1 ? bcol : bst1 - 1)];
      s_mask[tid] = (bcol < bst1) ? 1.0f : 0.0f;
    }
    __syncthreads();
    {
      const int tok = s_tok[ob];
#pragma unroll
      for (int i = 0; i < 8; ++i) {
        const int n = onb + i;
        s_theta[ob][n] = angle_emb[tok * DIM + n];
        s_accum[ob][n] = 0.0f;
      }
    }
    __syncthreads();

    // ---- layers ----
#pragma unroll
    for (int l = 0; l < DEPTH; ++l) {
      // o = 1 + scc*(-1 - ba + cos(d) + ba*cos(2d)), d = theta - accum - hist
#pragma unroll
      for (int t8 = 0; t8 < 8; ++t8) {
        const int f = 16 * t8 + lane_lo;
        const float ba = p_ba[l][t8], sc = p_scc[l][t8];
#pragma unroll
        for (int r = 0; r < 8; ++r) {
          const int b = r + 8 * half;
          const float d  = s_theta[b][f] - s_accum[b][f] - hist[l][t8][r];
          const float c1 = __cosf(d);
          const float c2 = 2.0f * c1 * c1 - 1.0f;  // cos(2d)
          s_o[h][b][f] = (bf16_t)(1.0f + sc * (-1.0f - ba + c1 + ba * c2));
        }
      }
      __syncthreads();

      // A fragments: 16x32 bf16 tiles from LDS (two ds_load_b128 each)
      v16bf afrag[4];
#pragma unroll
      for (int kc = 0; kc < 4; ++kc) {
        const bf16_t* rp = &s_o[h][lane_lo][kc * 32 + half * 8];
        const v8bf lo = *(const v8bf*)rp;
        const v8bf hi = *(const v8bf*)(rp + 16);
        afrag[kc] = __builtin_shufflevector(lo, hi, 0, 1, 2, 3, 4, 5, 6, 7,
                                            8, 9, 10, 11, 12, 13, 14, 15);
      }

      // raw = o @ Theta[l,h] : 8 N-tiles x 4 K-chunks of v_wmma_f32_16x16x32_bf16.
      // B-fragments stream from L2-resident thT every iteration; the volatile
      // LDS read of s_zero makes the addresses loop-variant so the compiler
      // cannot hoist 96 fragments out of the t-loop and spill them to scratch.
      const int z = *((volatile const int*)&s_zero);   // always 0
      v8f craw[8];
      const bf16_t* tb = thT + (size_t)((l * HEADS + h) * DIM * DIM + z);
#pragma unroll
      for (int nt = 0; nt < 8; ++nt) {
        v8f c;
#pragma unroll
        for (int i = 0; i < 8; ++i) c[i] = 0.0f;
        const bf16_t* col = tb + (size_t)(nt * 16 + lane_lo) * DIM + half * 8;
#pragma unroll
        for (int kc = 0; kc < 4; ++kc) {
          const v8bf lo = *(const v8bf*)(col + kc * 32);
          const v8bf hi = *(const v8bf*)(col + kc * 32 + 16);
          const v16bf bfrag = __builtin_shufflevector(
              lo, hi, 0, 1, 2, 3, 4, 5, 6, 7, 8, 9, 10, 11, 12, 13, 14, 15);
          c = __builtin_amdgcn_wmma_f32_16x16x32_bf16(
              false, afrag[kc], false, bfrag, (short)0, c, false, false);
        }
        craw[nt] = c;
      }

      // LayerNorm over n (per row b): partials per VGPR row, then 16-lane
      // butterfly reduction (masks 1..8 stay within each half).
      float sum[8], sq[8];
#pragma unroll
      for (int r = 0; r < 8; ++r) { sum[r] = 0.0f; sq[r] = 0.0f; }
#pragma unroll
      for (int nt = 0; nt < 8; ++nt)
#pragma unroll
        for (int r = 0; r < 8; ++r) {
          const float v = craw[nt][r];
          sum[r] += v; sq[r] += v * v;
        }
#pragma unroll
      for (int m = 1; m < 16; m <<= 1)
#pragma unroll
        for (int r = 0; r < 8; ++r) {
          sum[r] += __shfl_xor(sum[r], m, 32);
          sq[r]  += __shfl_xor(sq[r],  m, 32);
        }
      float mean[8], rstd[8];
#pragma unroll
      for (int r = 0; r < 8; ++r) {
        const float mu  = sum[r] * (1.0f / DIM);
        const float var = sq[r] * (1.0f / DIM) - mu * mu;
        mean[r] = mu;
        rstd[r] = rsqrtf(var + LN_EPS);
      }

      // normalize; hist <- h - head_clocks; accum += sum over heads (ds_add_f32)
#pragma unroll
      for (int nt = 0; nt < 8; ++nt) {
        const float w = p_lnw[l][nt], bb = p_lnb[l][nt];
        const int f = 16 * nt + lane_lo;
#pragma unroll
        for (int r = 0; r < 8; ++r) {
          const float hval = (craw[nt][r] - mean[r]) * rstd[r] * w + bb;
          hist[l][nt][r] = hval - hclk[nt];
          atomicAdd(&s_accum[r + 8 * half][f], hval);
        }
      }
      __syncthreads();
    }

    // ---- outputs: 4 planes of (B, dim), masked ----
    {
      const float tf = (float)t;
      const int bcol = bbase + ob;
      const int tok = s_tok[ob], lab = s_lab[ob];
      const float mf = s_mask[ob];
      const size_t plane = (size_t)TM * BTOT * DIM;
#pragma unroll
      for (int i = 0; i < 8; ++i) {
        const int n = onb + i;
        const float ang = tf * (tang[i] + s_theta[ob][n]) + s_accum[ob][n];
        const float cc = __cosf(ang);
        const float sn = __sinf(ang);
        const float er = emb_real[tok * DIM + n];
        const float ei = emb_imag[tok * DIM + n];
        const float lr = emb_real[lab * DIM + n];
        const float li = emb_imag[lab * DIM + n];
        const size_t base = ((size_t)t * BTOT + bcol) * DIM + n;
        __builtin_nontemporal_store((er * cc - ei * sn) * mf, out + base);
        __builtin_nontemporal_store((er * sn + ei * cc) * mf, out + plane + base);
        __builtin_nontemporal_store(lr * mf, out + 2 * plane + base);
        __builtin_nontemporal_store(li * mf, out + 3 * plane + base);
      }
    }
    __syncthreads();
  }
}

extern "C" void kernel_launch(void* const* d_in, const int* in_sizes, int n_in,
                              void* d_out, int out_size, void* d_ws, size_t ws_size,
                              hipStream_t stream) {
  const int*   data      = (const int*)  d_in[0];
  const int*   bs        = (const int*)  d_in[1];
  const float* angle_emb = (const float*)d_in[2];
  const float* emb_real  = (const float*)d_in[3];
  const float* emb_imag  = (const float*)d_in[4];
  const float* prop      = (const float*)d_in[5];
  const float* lima      = (const float*)d_in[6];
  const float* Theta     = (const float*)d_in[7];
  const float* ln_w      = (const float*)d_in[8];
  const float* ln_b      = (const float*)d_in[9];
  const float* init_h    = (const float*)d_in[10];
  const float* hcd       = (const float*)d_in[11];
  const float* rden      = (const float*)d_in[12];

  int*    off = (int*)d_ws;                       // 257 ints
  bf16_t* thT = (bf16_t*)((char*)d_ws + 2048);    // 3*8*128*128 bf16 = 768 KB

  prep_kernel<<<256, 256, 0, stream>>>(bs, Theta, off, thT);
  rotator_kernel<<<NWG, 256, 0, stream>>>(data, bs, angle_emb, emb_real, emb_imag,
                                          prop, lima, ln_w, ln_b, init_h, hcd, rden,
                                          off, thT, (float*)d_out);
}
